// NTM_45724221833216
// MI455X (gfx1250) — compile-verified, matrix operand order checked by reading, not run on hardware
//
#include <hip/hip_runtime.h>
#include <hip/hip_bf16.h>
#include <cstdint>
#include <cstddef>

// ---------------------------------------------------------------------------
// NTM forward, persistent kernel for gfx1250 (MI455X).
//   grid = 4 workgroups (16 batch rows each), block = 512 threads = 16 wave32.
//   LSTM/head/proj matmuls via v_wmma_f32_16x16x32_bf16 (M=16 batch tile).
//   c-state lives in registers in WMMA D-fragment layout; h in LDS as bf16 A.
//   A (LDS) and B (global) fragments are both double-buffered so WMMAs overlap
//   both DS and L2 latency; all tile addresses are immediate offsets from ONE
//   lane pointer per stream to stay under 256 VGPRs.
// ---------------------------------------------------------------------------

typedef __bf16 bf16;
typedef __attribute__((ext_vector_type(16))) __bf16        v16bf;
typedef __attribute__((ext_vector_type(8)))  float         v8f;
typedef __attribute__((ext_vector_type(8)))  unsigned int  u32x8;

#define B_    64
#define L_    1024
#define E_    256
#define H_    512
#define NSL   128
#define WD    64
#define NHD   4
#define O_    256
#define CIN_  512
#define TOT_  768
#define EPSF  1e-5f

#define ROWS    16
#define THREADS 512
#define NBLK    4

// ---- wave32 helpers -------------------------------------------------------
__device__ __forceinline__ float wave_sum(float v) {
#pragma unroll
  for (int i = 16; i >= 1; i >>= 1) v += __shfl_xor(v, i, 32);
  return v;
}
__device__ __forceinline__ float wave_max(float v) {
#pragma unroll
  for (int i = 16; i >= 1; i >>= 1) v = fmaxf(v, __shfl_xor(v, i, 32));
  return v;
}
__device__ __forceinline__ float grp8_sum(float v) {   // 8-lane aligned groups
#pragma unroll
  for (int i = 4; i >= 1; i >>= 1) v += __shfl_xor(v, i, 32);
  return v;
}
__device__ __forceinline__ float sigmoidf_(float x) { return 1.0f / (1.0f + __expf(-x)); }
__device__ __forceinline__ v8f zero8() {
  v8f z;
#pragma unroll
  for (int i = 0; i < 8; ++i) z[i] = 0.0f;
  return z;
}

// ---- WMMA fragment loader -------------------------------------------------
// laneptr already encodes the lane's (row, khalf) position; eoff is a
// compile-time-constant element offset so loads fold into immediate offsets.
// Per-lane data: 8x u32 = the 16-bit 16x32 fragment layout
//   lanes 0-15:  K pairs {0..7} in regs 0-3, {16..23} in regs 4-7
//   lanes 16-31: K pairs {8..15} in regs 0-3, {24..31} in regs 4-7
__device__ __forceinline__ v16bf load_frag(const bf16* laneptr, int eoff) {
  u32x8 u;
#pragma unroll
  for (int i = 0; i < 8; ++i) {
    const int k = ((i >> 2) << 4) + ((i & 3) << 1);
    u[i] = *(const unsigned int*)(laneptr + eoff + k);
  }
  return __builtin_bit_cast(v16bf, u);
}
__device__ __forceinline__ v8f wmma_bf16(v16bf a, v16bf b, v8f c) {
  return __builtin_amdgcn_wmma_f32_16x16x32_bf16(false, a, false, b, (short)0, c, false, false);
}

// ---- parameter block ------------------------------------------------------
struct NTMParams {
  const float* __restrict__ x_emb;
  const float* __restrict__ ln_in_g;  const float* __restrict__ ln_in_b;
  const float* __restrict__ b_ih;     const float* __restrict__ b_hh;
  const float* __restrict__ ln_ctrl_g; const float* __restrict__ ln_ctrl_b;
  const float* __restrict__ b_rk;     const float* __restrict__ b_wk;
  const float* __restrict__ W_ws;     const float* __restrict__ b_ws;
  const float* __restrict__ b_ev;
  const float* __restrict__ ln_rk_g;  const float* __restrict__ ln_rk_b;
  const float* __restrict__ ln_wk_g;  const float* __restrict__ ln_wk_b;
  const float* __restrict__ ln_mem_g; const float* __restrict__ ln_mem_b;
  const float* __restrict__ ln_out_g; const float* __restrict__ ln_out_b;
  const float* __restrict__ b_proj;
  const bf16* __restrict__ Wih; const bf16* __restrict__ Whh; const bf16* __restrict__ Wrk;
  const bf16* __restrict__ Wwk; const bf16* __restrict__ Wev; const bf16* __restrict__ Wproj;
  float* __restrict__ mem_ws;
  float* __restrict__ out_logits; float* __restrict__ out_mem;
  float* __restrict__ out_h;      float* __restrict__ out_c;
};

// ---- prep kernels ---------------------------------------------------------
__global__ void k_f32_to_bf16(const float* __restrict__ src, bf16* __restrict__ dst, int n) {
  int i = blockIdx.x * blockDim.x + threadIdx.x;
  if (i < n) dst[i] = (bf16)src[i];
}
__global__ void k_zero_f32(float* __restrict__ p, int n) {
  int i = blockIdx.x * blockDim.x + threadIdx.x;
  if (i < n) p[i] = 0.0f;
}

// ---- main persistent kernel ----------------------------------------------
__global__ __launch_bounds__(THREADS, 1)
void ntm_persistent(NTMParams p) {
  extern __shared__ char smem[];
  bf16*  s_ci  = (bf16*)smem;               // [16][512] LN'd input concat (A)
  bf16*  s_h   = s_ci + ROWS * CIN_;        // [16][512] raw h (A for W_hh)
  bf16*  s_out = s_h  + ROWS * H_;          // [16][768] hn | rc_new (A)
  float* s_rk  = (float*)(s_out + ROWS * TOT_);   // [16][256]
  float* s_wk  = s_rk + ROWS * 256;
  float* s_ev  = s_wk + ROWS * 256;
  float* s_rc  = s_ev + ROWS * 256;               // [16][256] read content f32
  float* s_ws  = s_rc + ROWS * 256;               // [16][4]

  const int tid  = threadIdx.x;
  const int lane = tid & 31;
  const int wave = tid >> 5;
  const int bt   = blockIdx.x;
  const int wcol0 = wave * 32;        // this wave's 32 LSTM columns
  const int nlo   = lane & 15;        // fragment row/col within tile
  const int khalf = (lane >> 4) << 3; // fragment K half select

  for (int i = tid; i < ROWS * H_;  i += THREADS) s_h[i]  = (bf16)0.0f;
  for (int i = tid; i < ROWS * 256; i += THREADS) s_rc[i] = 0.0f;
  float creg[16], hreg[16];
#pragma unroll
  for (int i = 0; i < 16; ++i) creg[i] = 0.0f;
  __syncthreads();

  for (int t = 0; t < L_; ++t) {
    // ---------------- Phase A: ci = LN(concat(x_t, rc)) -> s_ci (bf16)
    {
      const int r = wave;
      const float* xr = p.x_emb + (((size_t)(bt * ROWS + r)) * L_ + t) * E_;
      if (t + 1 < L_) __builtin_prefetch(xr + E_, 0, 1);
      float v[16]; float sm = 0.0f;
#pragma unroll
      for (int j = 0; j < 16; ++j) {
        const int idx = lane + 32 * j;
        const float x = (idx < E_) ? xr[idx] : s_rc[r * 256 + (idx - E_)];
        v[j] = x; sm += x;
      }
      sm = wave_sum(sm);
      const float mean = sm * (1.0f / CIN_);
      float var = 0.0f;
#pragma unroll
      for (int j = 0; j < 16; ++j) { const float d = v[j] - mean; var += d * d; }
      var = wave_sum(var);
      const float rstd = rsqrtf(var * (1.0f / CIN_) + EPSF);
#pragma unroll
      for (int j = 0; j < 16; ++j) {
        const int idx = lane + 32 * j;
        s_ci[r * CIN_ + idx] = (bf16)((v[j] - mean) * rstd * p.ln_in_g[idx] + p.ln_in_b[idx]);
      }
    }
    __syncthreads();

    // ---------------- Phase B: LSTM gates via WMMA (pipelined); c,h update
    // tile q = 0..7 -> gate g = q>>1, tn = q&1; element offset from one lane
    // pointer: q_off = g*512*512 + tn*16*512 (constant -> immediate offsets)
    {
      v8f acc[8];
#pragma unroll
      for (int q = 0; q < 8; ++q) acc[q] = zero8();
#pragma unroll
      for (int src = 0; src < 2; ++src) {
        const bf16* wl = (src ? p.Whh : p.Wih) + (size_t)(wcol0 + nlo) * 512 + khalf;
        const bf16* al = (src ? s_h : s_ci) + nlo * 512 + khalf;
        v16bf bcur[4], bnxt[4];
#pragma unroll
        for (int q = 0; q < 4; ++q)
          bcur[q] = load_frag(wl, (q >> 1) * 262144 + (q & 1) * 8192);
        v16bf a = load_frag(al, 0);
        for (int kk = 0; kk < 16; ++kk) {
          // prefetch next k-step's A fragment (hides LDS latency next iter)
          v16bf an = a;
          if (kk < 15) an = load_frag(al + 32, 0);
          // issue second half-group B loads before consuming first half-group
#pragma unroll
          for (int q = 4; q < 8; ++q)
            bnxt[q - 4] = load_frag(wl, (q >> 1) * 262144 + (q & 1) * 8192);
#pragma unroll
          for (int q = 0; q < 4; ++q)
            acc[q] = wmma_bf16(a, bcur[q], acc[q]);
          wl += 32;                       // next K chunk (shared by all tiles)
          // issue next k-step's first half-group while second half runs
          if (kk < 15) {
#pragma unroll
            for (int q = 0; q < 4; ++q)
              bcur[q] = load_frag(wl, (q >> 1) * 262144 + (q & 1) * 8192);
          }
#pragma unroll
          for (int q = 4; q < 8; ++q)
            acc[q] = wmma_bf16(a, bnxt[q - 4], acc[q]);
          a = an;
          al += 32;
        }
      }
#pragma unroll
      for (int tn = 0; tn < 2; ++tn) {
        const int n = wcol0 + tn * 16 + nlo;
        const float bi  = p.b_ih[n]          + p.b_hh[n];
        const float bff = p.b_ih[512 + n]    + p.b_hh[512 + n];
        const float bgg = p.b_ih[1024 + n]   + p.b_hh[1024 + n];
        const float boo = p.b_ih[1536 + n]   + p.b_hh[1536 + n];
#pragma unroll
        for (int j = 0; j < 8; ++j) {
          const float ig = sigmoidf_(acc[0 + tn][j] + bi);
          const float fg = sigmoidf_(acc[2 + tn][j] + bff);
          const float gg = tanhf(acc[4 + tn][j] + bgg);
          const float og = sigmoidf_(acc[6 + tn][j] + boo);
          const float cc = fg * creg[tn * 8 + j] + ig * gg;
          creg[tn * 8 + j] = cc;
          hreg[tn * 8 + j] = og * tanhf(cc);
        }
      }
    }
    __syncthreads();                    // all reads of previous h complete
    {
#pragma unroll
      for (int tn = 0; tn < 2; ++tn) {
        const int n = wcol0 + tn * 16 + nlo;
#pragma unroll
        for (int j = 0; j < 8; ++j) {
          const int m = ((lane >> 4) << 3) + j;     // D-fragment row
          s_h[m * H_ + n] = (bf16)hreg[tn * 8 + j];
        }
      }
    }
    __syncthreads();

    // ---------------- Phase C: hn = LN(h) -> s_out[:, 0:512]
    {
      const int r = wave;
      float v[16]; float sm = 0.0f;
#pragma unroll
      for (int j = 0; j < 16; ++j) {
        const int idx = lane + 32 * j;
        const float x = (float)s_h[r * H_ + idx];
        v[j] = x; sm += x;
      }
      sm = wave_sum(sm);
      const float mean = sm * (1.0f / H_);
      float var = 0.0f;
#pragma unroll
      for (int j = 0; j < 16; ++j) { const float d = v[j] - mean; var += d * d; }
      var = wave_sum(var);
      const float rstd = rsqrtf(var * (1.0f / H_) + EPSF);
#pragma unroll
      for (int j = 0; j < 16; ++j) {
        const int idx = lane + 32 * j;
        s_out[r * TOT_ + idx] = (bf16)((v[j] - mean) * rstd * p.ln_ctrl_g[idx] + p.ln_ctrl_b[idx]);
      }
    }
    __syncthreads();

    // ---------------- Phase D: head projections (rk/wk/ev via WMMA, ws VALU)
    {
#pragma unroll
      for (int i = 0; i < 3; ++i) {
        const int tt = wave * 3 + i;                 // 0..47 tiles
        const int mat = tt >> 4;                     // 0=rk 1=wk 2=ev
        const int ncol0 = (tt & 15) * 16;
        const bf16* W = (mat == 0) ? p.Wrk : (mat == 1) ? p.Wwk : p.Wev;
        const bf16* wl = W + (size_t)(ncol0 + nlo) * 512 + khalf;
        const bf16* al = s_out + nlo * TOT_ + khalf;
        v8f acc = zero8();
        v16bf bb = load_frag(wl, 0);
        v16bf aa = load_frag(al, 0);
        for (int kk = 0; kk < 16; ++kk) {
          v16bf bn = bb, an = aa;
          if (kk < 15) {                              // prefetch next K chunk
            bn = load_frag(wl, 32);
            an = load_frag(al + 32, 0);
          }
          acc = wmma_bf16(aa, bb, acc);
          bb = bn; aa = an;
          wl += 32; al += 32;
        }
        const int n = ncol0 + nlo;
        const float bias = (mat == 0) ? p.b_rk[n] : (mat == 1) ? p.b_wk[n] : p.b_ev[n];
#pragma unroll
        for (int j = 0; j < 8; ++j) {
          const int m = ((lane >> 4) << 3) + j;
          const float v = acc[j] + bias;
          if (mat == 0)      s_rk[m * 256 + n] = v;
          else if (mat == 1) s_wk[m * 256 + n] = v;
          else               s_ev[m * 256 + n] = sigmoidf_(v);
        }
      }
      if (tid < 64) {                                 // ws: 16 rows x 4 heads
        const int r = tid >> 2, n2 = tid & 3;
        float s = p.b_ws[n2];
        const float* wrow = p.W_ws + n2 * H_;
        for (int k = 0; k < H_; ++k) s += (float)s_out[r * TOT_ + k] * wrow[k];
        s_ws[r * 4 + n2] = sigmoidf_(s);
      }
    }
    __syncthreads();

    // ---------------- Phase E: LN rk / wk per 64-wide head word
    {
      const int g = tid >> 3;                 // 64 (row,head) groups
      const int r = g >> 2, hh = g & 3, sub = tid & 7;
      {
        float* ptr = s_rk + r * 256 + hh * 64;
        float v[8]; float sm = 0.0f;
#pragma unroll
        for (int j = 0; j < 8; ++j) { v[j] = ptr[sub * 8 + j]; sm += v[j]; }
        sm = grp8_sum(sm);
        const float mean = sm * (1.0f / WD);
        float var = 0.0f;
#pragma unroll
        for (int j = 0; j < 8; ++j) { const float d = v[j] - mean; var += d * d; }
        var = grp8_sum(var);
        const float rstd = rsqrtf(var * (1.0f / WD) + EPSF);
#pragma unroll
        for (int j = 0; j < 8; ++j) {
          const int w = sub * 8 + j;
          ptr[w] = (v[j] - mean) * rstd * p.ln_rk_g[w] + p.ln_rk_b[w];
        }
      }
      {
        float* ptr = s_wk + r * 256 + hh * 64;
        float v[8]; float sm = 0.0f;
#pragma unroll
        for (int j = 0; j < 8; ++j) { v[j] = ptr[sub * 8 + j]; sm += v[j]; }
        sm = grp8_sum(sm);
        const float mean = sm * (1.0f / WD);
        float var = 0.0f;
#pragma unroll
        for (int j = 0; j < 8; ++j) { const float d = v[j] - mean; var += d * d; }
        var = grp8_sum(var);
        const float rstd = rsqrtf(var * (1.0f / WD) + EPSF);
#pragma unroll
        for (int j = 0; j < 8; ++j) {
          const int w = sub * 8 + j;
          ptr[w] = (v[j] - mean) * rstd * p.ln_wk_g[w] + p.ln_wk_b[w];
        }
      }
    }
    __syncthreads();

    // ---------------- Phase F: memory addressing + read + erase/add
    {
      const int r = wave;                       // one wave per batch row
      float* memr = p.mem_ws + ((size_t)(bt * ROWS + r)) * NSL * WD;
      float rkh[4][2], wkh[4][2], evh[4][2], wsv[4];
#pragma unroll
      for (int h = 0; h < 4; ++h) {
        rkh[h][0] = s_rk[r * 256 + h * 64 + 2 * lane];
        rkh[h][1] = s_rk[r * 256 + h * 64 + 2 * lane + 1];
        wkh[h][0] = s_wk[r * 256 + h * 64 + 2 * lane];
        wkh[h][1] = s_wk[r * 256 + h * 64 + 2 * lane + 1];
        evh[h][0] = s_ev[r * 256 + h * 64 + 2 * lane];
        evh[h][1] = s_ev[r * 256 + h * 64 + 2 * lane + 1];
        wsv[h]    = s_ws[r * 4 + h];
      }
      const float gm0 = p.ln_mem_g[2 * lane], gm1 = p.ln_mem_g[2 * lane + 1];
      const float bm0 = p.ln_mem_b[2 * lane], bm1 = p.ln_mem_b[2 * lane + 1];

      float rsc[4][4], wsc[4][4];               // slot m = mi*32 + lane
#pragma unroll
      for (int mi = 0; mi < 4; ++mi) {
        for (int ml = 0; ml < 32; ++ml) {
          const int m = mi * 32 + ml;
          const float2 vv = *(const float2*)(memr + m * WD + 2 * lane);
          const float mean = wave_sum(vv.x + vv.y) * (1.0f / WD);
          const float d0 = vv.x - mean, d1 = vv.y - mean;
          const float rstd = rsqrtf(wave_sum(d0 * d0 + d1 * d1) * (1.0f / WD) + EPSF);
          const float n0 = d0 * rstd * gm0 + bm0;
          const float n1 = d1 * rstd * gm1 + bm1;
#pragma unroll
          for (int h = 0; h < 4; ++h) {
            const float pr = wave_sum(n0 * rkh[h][0] + n1 * rkh[h][1]);
            const float pw = wave_sum(n0 * wkh[h][0] + n1 * wkh[h][1]);
            if (lane == ml) { rsc[h][mi] = pr; wsc[h][mi] = pw; }
          }
        }
      }
      // softmax over 128 slots per head
#pragma unroll
      for (int h = 0; h < 4; ++h) {
        float mx = fmaxf(fmaxf(rsc[h][0], rsc[h][1]), fmaxf(rsc[h][2], rsc[h][3]));
        mx = wave_max(mx);
        float s = 0.0f;
#pragma unroll
        for (int mi = 0; mi < 4; ++mi) { rsc[h][mi] = __expf(rsc[h][mi] - mx); s += rsc[h][mi]; }
        s = wave_sum(s); const float inv = 1.0f / s;
#pragma unroll
        for (int mi = 0; mi < 4; ++mi) rsc[h][mi] *= inv;

        float mw = fmaxf(fmaxf(wsc[h][0], wsc[h][1]), fmaxf(wsc[h][2], wsc[h][3]));
        mw = wave_max(mw);
        float sw = 0.0f;
#pragma unroll
        for (int mi = 0; mi < 4; ++mi) { wsc[h][mi] = __expf(wsc[h][mi] - mw); sw += wsc[h][mi]; }
        sw = wave_sum(sw); const float invw = 1.0f / sw;
#pragma unroll
        for (int mi = 0; mi < 4; ++mi) wsc[h][mi] *= invw;
      }
      // read (old mem) + erase/add update in one pass
      float rcacc[4][2], adk[4][2];
#pragma unroll
      for (int h = 0; h < 4; ++h) {
        rcacc[h][0] = 0.0f; rcacc[h][1] = 0.0f;
        adk[h][0] = wsv[h] * wkh[h][0]; adk[h][1] = wsv[h] * wkh[h][1];
      }
#pragma unroll
      for (int mi = 0; mi < 4; ++mi) {
        for (int ml = 0; ml < 32; ++ml) {
          const int m = mi * 32 + ml;
          float rwv[4], wwv[4];
#pragma unroll
          for (int h = 0; h < 4; ++h) {
            rwv[h] = __shfl(rsc[h][mi], ml, 32);
            wwv[h] = __shfl(wsc[h][mi], ml, 32);
          }
          float2 vv = *(const float2*)(memr + m * WD + 2 * lane);
          float e0 = 0.0f, e1 = 0.0f, a0 = 0.0f, a1 = 0.0f;
#pragma unroll
          for (int h = 0; h < 4; ++h) {
            rcacc[h][0] += rwv[h] * vv.x; rcacc[h][1] += rwv[h] * vv.y;
            e0 += wwv[h] * evh[h][0];     e1 += wwv[h] * evh[h][1];
            a0 += wwv[h] * adk[h][0];     a1 += wwv[h] * adk[h][1];
          }
          float2 nv; nv.x = vv.x * (1.0f - e0) + a0; nv.y = vv.y * (1.0f - e1) + a1;
          *(float2*)(memr + m * WD + 2 * lane) = nv;
        }
      }
#pragma unroll
      for (int h = 0; h < 4; ++h) {
        s_rc[r * 256 + h * 64 + 2 * lane]     = rcacc[h][0];
        s_rc[r * 256 + h * 64 + 2 * lane + 1] = rcacc[h][1];
        s_out[r * TOT_ + 512 + h * 64 + 2 * lane]     = (bf16)rcacc[h][0];
        s_out[r * TOT_ + 512 + h * 64 + 2 * lane + 1] = (bf16)rcacc[h][1];
      }
    }
    __syncthreads();

    // ---------------- Phase G: out-LN over 768 (in place)
    {
      const int r = wave;
      float v[24]; float sm = 0.0f;
#pragma unroll
      for (int j = 0; j < 24; ++j) {
        const int idx = lane + 32 * j;
        const float x = (float)s_out[r * TOT_ + idx];
        v[j] = x; sm += x;
      }
      sm = wave_sum(sm);
      const float mean = sm * (1.0f / TOT_);
      float var = 0.0f;
#pragma unroll
      for (int j = 0; j < 24; ++j) { const float d = v[j] - mean; var += d * d; }
      var = wave_sum(var);
      const float rstd = rsqrtf(var * (1.0f / TOT_) + EPSF);
#pragma unroll
      for (int j = 0; j < 24; ++j) {
        const int idx = lane + 32 * j;
        s_out[r * TOT_ + idx] = (bf16)((v[j] - mean) * rstd * p.ln_out_g[idx] + p.ln_out_b[idx]);
      }
    }
    __syncthreads();

    // ---------------- Phase H: logits = out @ Wprojᵀ + b_proj
    {
      const int ncol0 = wave * 16;              // 16 waves x 16 cols = 256
      const bf16* wl = p.Wproj + (size_t)(ncol0 + nlo) * TOT_ + khalf;
      const bf16* al = s_out + nlo * TOT_ + khalf;
      v8f acc = zero8();
      v16bf bb = load_frag(wl, 0);
      v16bf aa = load_frag(al, 0);
      for (int kk = 0; kk < 24; ++kk) {
        v16bf bn = bb, an = aa;
        if (kk < 23) {                          // prefetch next K chunk
          bn = load_frag(wl, 32);
          an = load_frag(al + 32, 0);
        }
        acc = wmma_bf16(aa, bb, acc);
        bb = bn; aa = an;
        wl += 32; al += 32;
      }
      const int n = ncol0 + nlo;
      const float bias = p.b_proj[n];
#pragma unroll
      for (int j = 0; j < 8; ++j) {
        const int m = ((lane >> 4) << 3) + j;
        const size_t bg = (size_t)(bt * ROWS + m);
        p.out_logits[(bg * L_ + t) * O_ + n] = acc[j] + bias;
      }
    }
    __syncthreads();
  }

  // ---------------- epilogue: final mem, h, c
  {
    const size_t base = (size_t)bt * ROWS * NSL * WD;
    for (int i = tid; i < ROWS * NSL * WD; i += THREADS)
      p.out_mem[base + i] = p.mem_ws[base + i];
    for (int i = tid; i < ROWS * H_; i += THREADS) {
      const int r = i >> 9, k = i & 511;
      p.out_h[(size_t)(bt * ROWS + r) * H_ + k] = (float)s_h[i];
    }
#pragma unroll
    for (int tn = 0; tn < 2; ++tn) {
      const int n = wcol0 + tn * 16 + nlo;
#pragma unroll
      for (int j = 0; j < 8; ++j) {
        const int m = ((lane >> 4) << 3) + j;
        p.out_c[(size_t)(bt * ROWS + m) * H_ + n] = creg[tn * 8 + j];
      }
    }
  }
}

// ---------------------------------------------------------------------------
extern "C" void kernel_launch(void* const* d_in, const int* in_sizes, int n_in,
                              void* d_out, int out_size, void* d_ws, size_t ws_size,
                              hipStream_t stream) {
  (void)in_sizes; (void)n_in; (void)out_size; (void)ws_size;

  const float* x_emb    = (const float*)d_in[0];
  const float* ln_in_g  = (const float*)d_in[1];
  const float* ln_in_b  = (const float*)d_in[2];
  const float* W_ih     = (const float*)d_in[3];
  const float* W_hh     = (const float*)d_in[4];
  const float* b_ih     = (const float*)d_in[5];
  const float* b_hh     = (const float*)d_in[6];
  const float* ln_ctrl_g= (const float*)d_in[7];
  const float* ln_ctrl_b= (const float*)d_in[8];
  const float* W_rk     = (const float*)d_in[9];
  const float* b_rk     = (const float*)d_in[10];
  const float* W_wk     = (const float*)d_in[11];
  const float* b_wk     = (const float*)d_in[12];
  const float* W_ws     = (const float*)d_in[13];
  const float* b_ws     = (const float*)d_in[14];
  const float* W_ev     = (const float*)d_in[15];
  const float* b_ev     = (const float*)d_in[16];
  const float* ln_rk_g  = (const float*)d_in[17];
  const float* ln_rk_b  = (const float*)d_in[18];
  const float* ln_wk_g  = (const float*)d_in[19];
  const float* ln_wk_b  = (const float*)d_in[20];
  const float* ln_mem_g = (const float*)d_in[21];
  const float* ln_mem_b = (const float*)d_in[22];
  const float* ln_out_g = (const float*)d_in[23];
  const float* ln_out_b = (const float*)d_in[24];
  const float* W_proj   = (const float*)d_in[25];
  const float* b_proj   = (const float*)d_in[26];

  // workspace: bf16 weights + fp32 memory state
  bf16* wih = (bf16*)d_ws;
  bf16* whh = wih + 2048 * 512;
  bf16* wrk = whh + 2048 * 512;
  bf16* wwk = wrk + 256 * 512;
  bf16* wev = wwk + 256 * 512;
  bf16* wpr = wev + 256 * 512;
  float* mem_ws = (float*)(wpr + 256 * 768);

  auto cvt = [&](const float* s, bf16* d, int n) {
    k_f32_to_bf16<<<(n + 255) / 256, 256, 0, stream>>>(s, d, n);
  };
  cvt(W_ih,   wih, 2048 * 512);
  cvt(W_hh,   whh, 2048 * 512);
  cvt(W_rk,   wrk, 256 * 512);
  cvt(W_wk,   wwk, 256 * 512);
  cvt(W_ev,   wev, 256 * 512);
  cvt(W_proj, wpr, 256 * 768);
  k_zero_f32<<<(B_ * NSL * WD + 255) / 256, 256, 0, stream>>>(mem_ws, B_ * NSL * WD);

  float* out = (float*)d_out;
  NTMParams p;
  p.x_emb = x_emb;
  p.ln_in_g = ln_in_g;   p.ln_in_b = ln_in_b;
  p.b_ih = b_ih;         p.b_hh = b_hh;
  p.ln_ctrl_g = ln_ctrl_g; p.ln_ctrl_b = ln_ctrl_b;
  p.b_rk = b_rk;         p.b_wk = b_wk;
  p.W_ws = W_ws;         p.b_ws = b_ws;
  p.b_ev = b_ev;
  p.ln_rk_g = ln_rk_g;   p.ln_rk_b = ln_rk_b;
  p.ln_wk_g = ln_wk_g;   p.ln_wk_b = ln_wk_b;
  p.ln_mem_g = ln_mem_g; p.ln_mem_b = ln_mem_b;
  p.ln_out_g = ln_out_g; p.ln_out_b = ln_out_b;
  p.b_proj = b_proj;
  p.Wih = wih; p.Whh = whh; p.Wrk = wrk; p.Wwk = wwk; p.Wev = wev; p.Wproj = wpr;
  p.mem_ws = mem_ws;
  p.out_logits = out;
  p.out_mem = out + (size_t)B_ * L_ * O_;
  p.out_h   = p.out_mem + (size_t)B_ * NSL * WD;
  p.out_c   = p.out_h + (size_t)B_ * H_;

  const size_t smem = (size_t)(ROWS * CIN_ + ROWS * H_ + ROWS * TOT_) * sizeof(bf16)
                    + (size_t)(ROWS * 256 * 4 + ROWS * 4) * sizeof(float);
  ntm_persistent<<<dim3(NBLK), dim3(THREADS), smem, stream>>>(p);
}
